// MOELayer_11819749999023
// MI455X (gfx1250) — matrix operand order repaired and checked by reading, not compiled, not run on hardware
//
#include <hip/hip_runtime.h>

typedef __attribute__((ext_vector_type(16))) _Float16 v16h;
typedef __attribute__((ext_vector_type(8)))  _Float16 v8h;
typedef __attribute__((ext_vector_type(4)))  _Float16 v4h;
typedef __attribute__((ext_vector_type(8)))  float    v8f;
typedef __attribute__((ext_vector_type(4)))  unsigned v4u;
typedef __attribute__((ext_vector_type(8)))  int      v8i;
typedef __attribute__((ext_vector_type(4)))  int      v4i;

#define TOK 8192
#define DD  1024
#define HH  4096
#define EE  8
#define MT  32      // tokens per FFN tile (two 16-row WMMA tiles)
#define APB 2080    // sA row pitch in BYTES: 2048 + 2x16B pad (TDM pad: 16B per 1KB)
#define HPB 1040    // sH row pitch in BYTES: 1024 + 16B pad (bank-conflict free)

#if __has_builtin(__builtin_amdgcn_tensor_load_to_lds)
#define USE_TDM 1
#else
#define USE_TDM 0
#endif

#define WMMA_F16(A, B, C) \
  __builtin_amdgcn_wmma_f32_16x16x32_f16(false, (A), false, (B), (short)0, (C), false, false)

// ---------------- f32 -> f16 bulk convert (x only; grid-stride, 4-wide) --------
__global__ __launch_bounds__(256)
void k_cvt_f16(const float* __restrict__ src, _Float16* __restrict__ dst, long n) {
  long i = (long)blockIdx.x * blockDim.x + threadIdx.x;
  long stride = (long)gridDim.x * blockDim.x;
  for (long j = i * 4; j < n; j += stride * 4) {
    float4 v = *(const float4*)(src + j);
    v4h h;
    h.x = (_Float16)v.x; h.y = (_Float16)v.y;
    h.z = (_Float16)v.z; h.w = (_Float16)v.w;
    *(v4h*)(dst + j) = h;
  }
}

// ------- weight swizzle: f32 row-major (K x N) -> f16 WMMA-B fragment-major ----
// lane = grp*16 + nl holds B[kb*32 + grp*8 + t + (t >= 8 ? 8 : 0)][nb*16 + nl], t = 0..15, stored as
// 16 contiguous halves (32B) per lane; k-tiles contiguous for streaming.
__global__ __launch_bounds__(256)
void k_swz_w(const float* __restrict__ src, _Float16* __restrict__ dst,
             int K, int N, int nexp) {
  const int KT = K >> 5, NT = N >> 4;
  const long nblk = (long)nexp * NT * KT * 32;
  long idx = (long)blockIdx.x * blockDim.x + threadIdx.x;
  const long stride = (long)gridDim.x * blockDim.x;
  for (; idx < nblk; idx += stride) {
    const int lane = (int)(idx & 31);
    long tl = idx >> 5;
    const int kb = (int)(tl % KT);
    long t2 = tl / KT;
    const int nb = (int)(t2 % NT);
    const int e  = (int)(t2 / NT);
    const int grp = lane >> 4, nl = lane & 15;
    const float* sp = src + (long)e * K * N + (long)(kb * 32 + grp * 8) * N + nb * 16 + nl;
    union { v16h v; _Float16 h[16]; } b;
#pragma unroll
    for (int t = 0; t < 16; ++t) b.h[t] = (_Float16)sp[(long)(t < 8 ? t : t + 8) * N];
    *(v16h*)(dst + idx * 16) = b.v;
  }
}

// ---------------- router: logits, top-2 softmax gates, expert token lists -------
__global__ __launch_bounds__(256)
void k_router(const float* __restrict__ x, const float* __restrict__ rw,
              const float* __restrict__ rb, float* __restrict__ gates,
              int* __restrict__ counts, int* __restrict__ lists) {
  const int wv = threadIdx.x >> 5;
  const int lane = threadIdx.x & 31;
  const int token = blockIdx.x * 8 + wv;   // one wave32 per token

  float acc[EE];
#pragma unroll
  for (int e = 0; e < EE; ++e) acc[e] = 0.f;
  const float* xr = x + (long)token * DD;
  for (int d = lane; d < DD; d += 32) {
    const float xv = xr[d];
#pragma unroll
    for (int e = 0; e < EE; ++e) acc[e] += xv * rw[d * EE + e];
  }
#pragma unroll
  for (int e = 0; e < EE; ++e) {
#pragma unroll
    for (int off = 16; off > 0; off >>= 1) acc[e] += __shfl_down(acc[e], off);
  }
  if (lane == 0) {
    float lg[EE];
#pragma unroll
    for (int e = 0; e < EE; ++e) lg[e] = acc[e] + rb[e];
    int e0 = 0;
#pragma unroll
    for (int e = 1; e < EE; ++e) if (lg[e] > lg[e0]) e0 = e;
    int e1 = (e0 == 0) ? 1 : 0;
#pragma unroll
    for (int e = 0; e < EE; ++e) if (e != e0 && lg[e] > lg[e1]) e1 = e;
    const float wv1 = __expf(lg[e1] - lg[e0]);   // softmax over the 2 selected
    const float s = 1.f + wv1;
    float* gr = gates + (long)token * EE;
#pragma unroll
    for (int e = 0; e < EE; ++e) gr[e] = 0.f;
    gr[e0] = 1.f / s;
    gr[e1] = wv1 / s;
    int p0 = atomicAdd(&counts[e0], 1); lists[e0 * TOK + p0] = token;
    int p1 = atomicAdd(&counts[e1], 1); lists[e1 * TOK + p1] = token;
  }
}

// ---------------- fused expert FFN: 32-token tile, GEMM1->relu->GEMM2 -----------
__global__ __launch_bounds__(256)
void k_moe_ffn(const _Float16* __restrict__ xh,
               const _Float16* __restrict__ w1s, const float* __restrict__ b1,
               const _Float16* __restrict__ w2s, const float* __restrict__ b2,
               const float* __restrict__ gates, const int* __restrict__ counts,
               const int* __restrict__ lists, float* __restrict__ out) {
  const int e = blockIdx.y;
  const int cnt = counts[e];
  const int tile = blockIdx.x;
  if (tile * MT >= cnt) return;
  const int rows = min(MT, cnt - tile * MT);

  extern __shared__ char smem[];
  char* sAb = smem;                              // [MT] rows @ APB pitch (f16 + pad)
  char* sHb = smem + MT * APB;                   // [MT] rows @ HPB pitch (f16 + pad)
  int*   sTok  = (int*)(sHb + MT * HPB);
  float* sGate = (float*)(sTok + MT);

  const int tid  = threadIdx.x;
  const int lane = tid & 31;
  const int wv   = tid >> 5;      // 0..7 waves
  const int ln16 = lane & 15;
  const int grp  = lane >> 4;     // half-wave group

  if (tid < MT) {
    int tok = 0; float g = 0.f;
    if (tid < rows) {
      tok = lists[e * TOK + tile * MT + tid];
      g = gates[(long)tok * EE + e];
    }
    sTok[tid] = tok; sGate[tid] = g;
  }
  __syncthreads();

#if USE_TDM
  // --- Tensor Data Mover gather: 2 x (16 scattered token rows x 1024 f16) ------
  // D# per ISA 8.x: gather_mode, 16-bit indices, data_size=2B, tile_dim0=1024,
  // pad 16B per 1KB so the LDS image has the APB (2080B) row pitch.
  if (wv == 0) {
    const int tokv = sTok[lane];
    unsigned pk[16];
#pragma unroll
    for (int j = 0; j < 16; ++j) {
      unsigned lo = ((unsigned)__shfl(tokv, 2 * j)) & 0xFFFFu;
      unsigned hi = ((unsigned)__shfl(tokv, 2 * j + 1)) & 0xFFFFu;
      pk[j] = __builtin_amdgcn_readfirstlane(lo | (hi << 16));
    }
    const unsigned long ga = (unsigned long)(const void*)xh;
    v4u g0 = { 1u | (1u << 31),                       // count=1, gather_mode, idx16
               0u,                                    // lds_addr: tile A
               (unsigned)(ga & 0xFFFFFFFFu),
               (unsigned)((ga >> 32) & 0x01FFFFFFu) | (2u << 30) };  // type=2
    v8i g1 = { (int)0x07D10000,                       // data_size=1(2B), pad_en,
                                                      // interval=7(1KB), amount=3(16B)
               (int)(DD << 16),                       // tensor_dim0 = 1024
               (int)(TOK << 16),                      // tensor_dim1 = 8192
               (int)(DD << 16),                       // tile_dim0 = 1024
               16,                                    // tile_dim1 = 16 gather rows
               DD,                                    // tensor_dim0_stride = 1024
               0, 0 };
    v4i g2 = { (int)pk[0], (int)pk[1], (int)pk[2], (int)pk[3] };
    v4i g3 = { (int)pk[4], (int)pk[5], (int)pk[6], (int)pk[7] };
#if __has_include(<hip/amd_detail/amd_gfx1250_TDM.h>)
    v8i gz = {};
    __builtin_amdgcn_tensor_load_to_lds(g0, g1, g2, g3, gz, 0);
#else
    __builtin_amdgcn_tensor_load_to_lds(g0, g1, g2, g3, 0);
#endif
    g0.y = 16u * APB;                                 // tile B: rows 16..31
    v4i g2b = { (int)pk[8],  (int)pk[9],  (int)pk[10], (int)pk[11] };
    v4i g3b = { (int)pk[12], (int)pk[13], (int)pk[14], (int)pk[15] };
#if __has_include(<hip/amd_detail/amd_gfx1250_TDM.h>)
    __builtin_amdgcn_tensor_load_to_lds(g0, g1, g2b, g3b, gz, 0);
#else
    __builtin_amdgcn_tensor_load_to_lds(g0, g1, g2b, g3b, 0);
#endif
    __builtin_amdgcn_s_wait_tensorcnt(0);
  }
#else
  // --- fallback: manual staging, same 2080B pitch (16B pad after each 1KB) -----
  {
    const int m = tid >> 3, sub = tid & 7;
    v8h* dstp = (v8h*)(sAb + m * APB + sub * 256 + ((sub >= 4) ? 16 : 0));
    if (m < rows) {
      const v8h* srcp = (const v8h*)(xh + (long)sTok[m] * DD + sub * 128);
#pragma unroll
      for (int i = 0; i < 16; ++i) dstp[i] = srcp[i];
    } else {
      v8h z = {};
#pragma unroll
      for (int i = 0; i < 16; ++i) dstp[i] = z;
    }
  }
#endif

  const int KT1 = DD >> 5;   // 32 k-tiles in w1
  const int KT2 = HH >> 5;   // 128 k-tiles in w2
  const _Float16* w1se = w1s + (long)e * DD * HH;
  const _Float16* w2se = w2s + (long)e * HH * DD;
  const float* b1e = b1 + (long)e * HH;
  const float* b2e = b2 + (long)e * DD;

  v8f accD[2][8] = {};   // 32x128 f32 output accumulator per wave, persistent

  for (int ch = 0; ch < 8; ++ch) {     // H in 8 chunks of 512
    __syncthreads();                   // sA ready / sH consumed -> rewritable
    // ---- GEMM1: h[:, ch*512 + wv*64 .. +64) = relu(x @ w1 + b1) ----
    {
      const int colbase = wv * 64;
      v8f c[2][4] = {};
      for (int k0 = 0; k0 < DD; k0 += 32) {
        const char* r0 = sAb + ln16 * APB + (k0 << 1) + ((k0 >> 9) << 4);
        const char* r1 = r0 + 16 * APB;
        union { v16h v; v4u q[2]; } a0, a1;
        a0.q[0] = *(const v4u*)(r0 + 16 * grp);
        a0.q[1] = *(const v4u*)(r0 + 32 + 16 * grp);
        a1.q[0] = *(const v4u*)(r1 + 16 * grp);
        a1.q[1] = *(const v4u*)(r1 + 32 + 16 * grp);
        const int kb = k0 >> 5;
#pragma unroll
        for (int tt = 0; tt < 4; ++tt) {
          const int nb = (ch * 512 + colbase + tt * 16) >> 4;
          const _Float16* bp = w1se + (((long)(nb * KT1 + kb) * 32 + lane) << 4);
          const v16h bv = *(const v16h*)bp;
          __builtin_prefetch(bp + 512, 0, 3);   // next k-tile (1KB ahead)
          c[0][tt] = WMMA_F16(a0.v, bv, c[0][tt]);
          c[1][tt] = WMMA_F16(a1.v, bv, c[1][tt]);
        }
      }
#pragma unroll
      for (int tt = 0; tt < 4; ++tt) {
        const int lc = colbase + tt * 16 + ln16;
        const float bias = b1e[ch * 512 + lc];
#pragma unroll
        for (int mt = 0; mt < 2; ++mt)
#pragma unroll
          for (int r = 0; r < 8; ++r) {
            const int m = mt * 16 + r + 8 * grp;
            float hv = c[mt][tt][r] + bias;
            hv = hv > 0.f ? hv : 0.f;
            *(_Float16*)(sHb + m * HPB + lc * 2) = (_Float16)hv;
          }
      }
    }
    __syncthreads();
    // ---- GEMM2 partial: acc += h_chunk @ w2[ch*512.., wv*128..] ----
    for (int k0 = 0; k0 < 512; k0 += 32) {
      const char* r0 = sHb + ln16 * HPB + (k0 << 1);
      const char* r1 = r0 + 16 * HPB;
      union { v16h v; v4u q[2]; } a0, a1;
      a0.q[0] = *(const v4u*)(r0 + 16 * grp);
      a0.q[1] = *(const v4u*)(r0 + 32 + 16 * grp);
      a1.q[0] = *(const v4u*)(r1 + 16 * grp);
      a1.q[1] = *(const v4u*)(r1 + 32 + 16 * grp);
      const int kb = (ch * 512 + k0) >> 5;
#pragma unroll
      for (int nt = 0; nt < 8; ++nt) {
        const int nb = wv * 8 + nt;
        const _Float16* bp = w2se + (((long)(nb * KT2 + kb) * 32 + lane) << 4);
        const v16h bv = *(const v16h*)bp;
        __builtin_prefetch(bp + 512, 0, 3);
        accD[0][nt] = WMMA_F16(a0.v, bv, accD[0][nt]);
        accD[1][nt] = WMMA_F16(a1.v, bv, accD[1][nt]);
      }
    }
  }

  // ---- combine: out[token] += gate * (acc + b2) ----
#pragma unroll
  for (int nt = 0; nt < 8; ++nt) {
    const int n = wv * 128 + nt * 16 + ln16;
    const float bias = b2e[n];
#pragma unroll
    for (int mt = 0; mt < 2; ++mt)
#pragma unroll
      for (int r = 0; r < 8; ++r) {
        const int m = mt * 16 + r + 8 * grp;
        if (m < rows) {
          const float val = sGate[m] * (accD[mt][nt][r] + bias);
          unsafeAtomicAdd(&out[(long)sTok[m] * DD + n], val);
        }
      }
  }
}

extern "C" void kernel_launch(void* const* d_in, const int* in_sizes, int n_in,
                              void* d_out, int out_size, void* d_ws, size_t ws_size,
                              hipStream_t stream) {
  (void)in_sizes; (void)n_in; (void)out_size; (void)ws_size;
  const float* x  = (const float*)d_in[0];
  const float* rw = (const float*)d_in[1];
  const float* rb = (const float*)d_in[2];
  const float* w1 = (const float*)d_in[3];
  const float* b1 = (const float*)d_in[4];
  const float* w2 = (const float*)d_in[5];
  const float* b2 = (const float*)d_in[6];
  float* out = (float*)d_out;

  char* ws = (char*)d_ws;
  size_t o = 0;
  float* gates = (float*)(ws + o);  o += (size_t)TOK * EE * 4;
  int* counts  = (int*)(ws + o);    o += 256;
  int* lists   = (int*)(ws + o);    o += (size_t)EE * TOK * 4;
  _Float16* xh  = (_Float16*)(ws + o); o += (size_t)TOK * DD * 2;
  _Float16* w1s = (_Float16*)(ws + o); o += (size_t)EE * DD * HH * 2;
  _Float16* w2s = (_Float16*)(ws + o); o += (size_t)EE * HH * DD * 2;

  hipMemsetAsync(counts, 0, 64, stream);
  hipMemsetAsync(out, 0, (size_t)TOK * DD * sizeof(float), stream);

  k_cvt_f16<<<2048, 256, 0, stream>>>(x, xh, (long)TOK * DD);
  k_swz_w<<<2048, 256, 0, stream>>>(w1, w1s, DD, HH, EE);
  k_swz_w<<<2048, 256, 0, stream>>>(w2, w2s, HH, DD, EE);

  k_router<<<TOK / 8, 256, 0, stream>>>(x, rw, rb, gates, counts, lists);

  const size_t shmem = (size_t)MT * APB + (size_t)MT * HPB + (size_t)MT * 8;
  dim3 grid(TOK / MT, EE);
  k_moe_ffn<<<grid, 256, shmem, stream>>>(xh, w1s, b1, w2s, b2,
                                          gates, counts, lists, out);
}
